// Text_Decoder_85916525789534
// MI455X (gfx1250) — compile-verified, matrix operand order checked by reading, not run on hardware
//
#include <hip/hip_runtime.h>
#include <hip/hip_bf16.h>

// Problem sizes (fixed by the reference)
#define B_  2048
#define E_  128
#define L_  77
#define W_  64
#define DI_ 128
#define DS_ 16
#define DC_ 4
#define DR_ 4

#define NB        16      // batch rows per workgroup (== WMMA M)
#define NTHREADS  1024    // 32 waves of 32

typedef __bf16 bf16_t;
typedef bf16_t v16bf __attribute__((ext_vector_type(16)));
typedef float  v8f   __attribute__((ext_vector_type(8)));
typedef float  v2f   __attribute__((ext_vector_type(2)));
typedef unsigned int v4u __attribute__((ext_vector_type(4)));
typedef int          v8i __attribute__((ext_vector_type(8)));
typedef int          v4i __attribute__((ext_vector_type(4)));

// ---------- scalar helpers ----------
__device__ __forceinline__ unsigned short f2bf(float a) {
  unsigned int u = __float_as_uint(a);
  return (unsigned short)((u + 0x7FFFu + ((u >> 16) & 1u)) >> 16);   // RNE
}
__device__ __forceinline__ unsigned int f2bf_pk(float a, float b) {
  return (unsigned int)f2bf(a) | ((unsigned int)f2bf(b) << 16);
}
__device__ __forceinline__ float bf2f(unsigned short h) {
  return __uint_as_float(((unsigned int)h) << 16);
}
__device__ __forceinline__ float sigmoid_(float x) { return 1.0f / (1.0f + __expf(-x)); }
__device__ __forceinline__ float softplus_(float x) {
  return (x > 20.0f) ? x : __logf(1.0f + __expf(x));
}

// ---------- WMMA 16x16x32 bf16 fragment gather from packed LDS ----------
// ISA 7.12.2 16-bit A layout; B mirrors A with row == output column.
__device__ __forceinline__ v16bf load_frag(const unsigned int* pk, int stride_u,
                                           int row0, int k0u) {
  const int lane = threadIdx.x & 31;
  const unsigned int* p = pk + (row0 + (lane & 15)) * stride_u + k0u + ((lane >> 4) << 2);
  union { uint4 q[2]; v16bf v; } x;
  x.q[0] = *(const uint4*)(p);
  x.q[1] = *(const uint4*)(p + 8);
  return x.v;
}

// ---------- WMMA 16x16x4 f32 fragment gather (32-bit A layout, ISA 7.12.2) ---
// lanes 0..15: rows, K = k0,k0+1 ; lanes 16..31: same rows, K = k0+2,k0+3
__device__ __forceinline__ v2f load_frag_f32(const float* base, int stride,
                                             int row0, int k0) {
  const int lane = threadIdx.x & 31;
  const float* p = base + (row0 + (lane & 15)) * stride + k0 + ((lane >> 4) << 1);
  return *(const v2f*)p;
}

// ---------- LDS ----------
struct SM {
  float        eTf[NB][E_];             // encoded_text tile fp32       8 KB
  float        decwf[W_][E_];           // TDM landing buffer (fp32)   32 KB
  float        db[W_];
  unsigned int inw[2 * DI_][W_ / 2];    // in_w bf16                   32 KB
  unsigned int xpw[48][DI_ / 2];        // xproj_w bf16 (padded)       12 KB
  unsigned int oww[W_][DI_ / 2];        // out_w bf16                  16 KB
  float        dtw[DI_][DR_];
  float        dtb[DI_];
  float        cw[DI_][DC_];
  float        cb[DI_];
  float        Dv[DI_];
  float        Af[DI_][DS_];            // A = -exp(A_log)              8 KB
  unsigned short Urh[DC_][NB][DI_];     // conv ring, bf16             16 KB
  // per-timestep scratch
  float          Xf[NB][W_];            // x_t fp32                     4 KB
  unsigned int   Xpk[NB][W_ / 2];       // x_t bf16 packed              2 KB
  unsigned short Zh[NB][DI_];           // gate z_t, bf16               4 KB
  unsigned int   Upk[NB][DI_ / 2];      // u_t bf16 packed              4 KB
  float          xdbl[NB][48];          // [dt(4)|B(16)|C(16)|pad]      3 KB
  unsigned int   Ypk[NB][DI_ / 2];      // gated scan output bf16       4 KB
};                                      // total ~151 KB

// ---------- TDM: DMA one contiguous 64x128 fp32 slice into LDS ----------
#if __has_builtin(__builtin_amdgcn_tensor_load_to_lds)
__device__ __forceinline__ void tdm_load_slice(const float* gsrc, void* ldst) {
  unsigned long long ga = (unsigned long long)(const void*)gsrc;
  unsigned int ldsa = (unsigned int)(size_t)ldst;
  // D# group 0: count=1 | lds_addr | global_addr[56:0] | type=2
  v4u g0;
  g0.x = 1u;
  g0.y = ldsa;
  g0.z = (unsigned int)(ga & 0xFFFFFFFFull);
  g0.w = (unsigned int)((ga >> 32) & 0x1FFFFFFull) | 0x80000000u;
  // D# group 1: data_size=4B, 2D tile 128x64, tensor_dim0=128,
  //             tensor_dim1 = L*W rows, row stride 128
  v8i g1;
  g1[0] = (int)(2u << 16);                                   // data_size=2 (4B)
  g1[1] = (int)(((unsigned)E_ & 0xFFFFu) << 16);             // tensor_dim0 lo16
  g1[2] = (int)((((unsigned)E_) >> 16) |
                (((unsigned)(L_ * W_) & 0xFFFFu) << 16));    // td0 hi | td1 lo
  g1[3] = (int)((((unsigned)(L_ * W_)) >> 16) |
                (((unsigned)E_) << 16));                     // td1 hi | tile_dim0
  g1[4] = (int)W_;                                           // tile_dim1 (tile_dim2=0)
  g1[5] = (int)E_;                                           // tensor_dim0_stride lo32
  g1[6] = 0; g1[7] = 0;
  v4i gz = {0, 0, 0, 0};
#if defined(__clang_major__) && (__clang_major__ >= 23)
  v8i gz8 = {0, 0, 0, 0, 0, 0, 0, 0};
  __builtin_amdgcn_tensor_load_to_lds(g0, g1, gz, gz, gz8, 0);
#else
  __builtin_amdgcn_tensor_load_to_lds(g0, g1, gz, gz, 0);
#endif
}
#elif defined(__HIP_DEVICE_COMPILE__)
// Device pass must have the TDM builtin — fail loudly so we know.
#error "gfx1250 tensor_load_to_lds builtin not available in device pass"
#else
// Host pass: never executed, just needs to parse.
__device__ __forceinline__ void tdm_load_slice(const float*, void*) {}
#endif

#if __has_builtin(__builtin_amdgcn_s_wait_tensorcnt) || defined(__HIP_DEVICE_COMPILE__)
#define WAIT_TENSORCNT0() __builtin_amdgcn_s_wait_tensorcnt(0)
#else
#define WAIT_TENSORCNT0() ((void)0)
#endif

__global__ __launch_bounds__(NTHREADS)
void mamba_text_decoder_fused(const float* __restrict__ enc,
                              const float* __restrict__ dec_w,
                              const float* __restrict__ dec_b,
                              const float* __restrict__ in_w,
                              const float* __restrict__ conv_w,
                              const float* __restrict__ conv_b,
                              const float* __restrict__ xproj_w,
                              const float* __restrict__ dtproj_w,
                              const float* __restrict__ dtproj_b,
                              const float* __restrict__ A_log,
                              const float* __restrict__ Dp,
                              const float* __restrict__ out_w,
                              float* __restrict__ out) {
  __shared__ SM sm;
  const int tid  = threadIdx.x;
  const int wave = tid >> 5;
  const int lane = tid & 31;
  const int b0   = blockIdx.x * NB;

  // ================= one-time staging =================
  for (int i = tid; i < 2 * DI_ * (W_ / 2); i += NTHREADS) {
    int n = i >> 5, k = i & 31;
    sm.inw[n][k] = f2bf_pk(in_w[n * W_ + 2 * k], in_w[n * W_ + 2 * k + 1]);
  }
  for (int i = tid; i < 48 * (DI_ / 2); i += NTHREADS) {
    int n = i >> 6, k = i & 63;
    float a = 0.f, b = 0.f;
    if (n < DR_ + 2 * DS_) { a = xproj_w[n * DI_ + 2 * k]; b = xproj_w[n * DI_ + 2 * k + 1]; }
    sm.xpw[n][k] = f2bf_pk(a, b);
  }
  for (int i = tid; i < W_ * (DI_ / 2); i += NTHREADS) {
    int n = i >> 6, k = i & 63;
    sm.oww[n][k] = f2bf_pk(out_w[n * DI_ + 2 * k], out_w[n * DI_ + 2 * k + 1]);
  }
  for (int i = tid; i < DI_ * DR_; i += NTHREADS) sm.dtw[i >> 2][i & 3] = dtproj_w[i];
  for (int i = tid; i < DI_ * DC_; i += NTHREADS) sm.cw[i >> 2][i & 3]  = conv_w[i];
  if (tid < DI_) { sm.dtb[tid] = dtproj_b[tid]; sm.cb[tid] = conv_b[tid]; sm.Dv[tid] = Dp[tid]; }
  for (int i = tid; i < DI_ * DS_; i += NTHREADS) sm.Af[i >> 4][i & 15] = -__expf(A_log[i]);
  for (int i = tid; i < NB * E_; i += NTHREADS) {
    int m = i >> 7, k = i & 127;
    sm.eTf[m][k] = enc[(b0 + m) * E_ + k];
  }
  // kick off the DMA for t=0 while everything else settles
  if (wave == 0) tdm_load_slice(dec_w, &sm.decwf[0][0]);
  __syncthreads();

  // ================= per-thread recurrent state (2 (b,d) pairs) ==========
  const int sb = tid >> 6;         // batch row in tile
  const int sj = tid & 63;         // d-pair index
  const int d0 = sj * 2;
  float h0[DS_], h1[DS_];
  #pragma unroll
  for (int s = 0; s < DS_; ++s) { h0[s] = 0.f; h1[s] = 0.f; }

  // ================= sequential scan over L ==============================
  for (int t = 0; t < L_; ++t) {
    // --- wait for this step's dec_w slice (DMA issued one step ahead) ---
    if (wave == 0) WAIT_TENSORCNT0();
    if (tid < W_) sm.db[tid] = dec_b[t * W_ + tid];
    if (t + 1 < L_ && tid < W_)                     // warm L2 for next slice
      __builtin_prefetch(dec_w + (size_t)((t + 1) * W_ + tid) * E_, 0, 0);
    __syncthreads();

    // --- dec GEMM in fp32: X[16,64] = eTf[16,128] @ decwf^T + db (waves 0..3)
    if (wave < 4) {
      v8f acc = {};
      #pragma unroll
      for (int kt = 0; kt < E_ / 4; ++kt) {
        v2f av = load_frag_f32(&sm.eTf[0][0],   E_, 0,         kt * 4);
        v2f bv = load_frag_f32(&sm.decwf[0][0], E_, wave * 16, kt * 4);
        acc = __builtin_amdgcn_wmma_f32_16x16x4_f32(false, av, false, bv,
                                                    (short)0, acc, false, false);
      }
      int n  = (wave << 4) + (lane & 15);
      int mb = (lane >> 4) * 8;
      float bias = sm.db[n];
      #pragma unroll
      for (int r = 0; r < 8; ++r) sm.Xf[mb + r][n] = acc[r] + bias;
    }
    __syncthreads();

    // --- landing buffer is free: start DMA for t+1; repack X to bf16 ---
    if (wave == 0 && t + 1 < L_)
      tdm_load_slice(dec_w + (size_t)(t + 1) * W_ * E_, &sm.decwf[0][0]);
    if (tid < NB * (W_ / 2)) {
      int m = tid >> 5, k = tid & 31;
      sm.Xpk[m][k] = f2bf_pk(sm.Xf[m][2 * k], sm.Xf[m][2 * k + 1]);
    }
    __syncthreads();

    // --- in_proj GEMM: XZ[16,256] = X[16,64] @ inw^T (waves 0..15) ---
    if (wave < 16) {
      v8f acc = {};
      #pragma unroll
      for (int kt = 0; kt < 2; ++kt) {
        v16bf av = load_frag(&sm.Xpk[0][0], W_ / 2, 0,         kt * 16);
        v16bf bv = load_frag(&sm.inw[0][0], W_ / 2, wave * 16, kt * 16);
        acc = __builtin_amdgcn_wmma_f32_16x16x32_bf16(false, av, false, bv,
                                                      (short)0, acc, false, false);
      }
      int n  = (wave << 4) + (lane & 15);
      int mb = (lane >> 4) * 8;
      int slot = t & 3;
      if (n < DI_) {
        #pragma unroll
        for (int r = 0; r < 8; ++r) sm.Urh[slot][mb + r][n] = f2bf(acc[r]);
      } else {
        #pragma unroll
        for (int r = 0; r < 8; ++r) sm.Zh[mb + r][n - DI_] = f2bf(acc[r]);
      }
    }
    __syncthreads();

    // --- causal depthwise conv (DC=4) + SiLU (all threads, 2 channels) ---
    float u0, u1;
    {
      float acc0 = sm.cb[d0], acc1 = sm.cb[d0 + 1];
      #pragma unroll
      for (int i = 0; i < DC_; ++i) {
        int tt = t - (DC_ - 1) + i;
        if (tt >= 0) {
          int sl = tt & 3;
          acc0 += sm.cw[d0][i]     * bf2f(sm.Urh[sl][sb][d0]);
          acc1 += sm.cw[d0 + 1][i] * bf2f(sm.Urh[sl][sb][d0 + 1]);
        }
      }
      u0 = acc0 * sigmoid_(acc0);
      u1 = acc1 * sigmoid_(acc1);
      sm.Upk[sb][sj] = f2bf_pk(u0, u1);
    }
    __syncthreads();

    // --- xproj GEMM: XDBL[16,48] = U[16,128] @ xpw^T (waves 0..2) ---
    if (wave < 3) {
      v8f acc = {};
      #pragma unroll
      for (int kt = 0; kt < 4; ++kt) {
        v16bf av = load_frag(&sm.Upk[0][0], DI_ / 2, 0,         kt * 16);
        v16bf bv = load_frag(&sm.xpw[0][0], DI_ / 2, wave * 16, kt * 16);
        acc = __builtin_amdgcn_wmma_f32_16x16x32_bf16(false, av, false, bv,
                                                      (short)0, acc, false, false);
      }
      int n  = (wave << 4) + (lane & 15);
      int mb = (lane >> 4) * 8;
      #pragma unroll
      for (int r = 0; r < 8; ++r) sm.xdbl[mb + r][n] = acc[r];
    }
    __syncthreads();

    // --- dt-proj + softplus, selective-scan update, skip + gate ---
    {
      float x0 = sm.xdbl[sb][0], x1 = sm.xdbl[sb][1];
      float x2 = sm.xdbl[sb][2], x3 = sm.xdbl[sb][3];
      float dt0 = softplus_(sm.dtb[d0] +
          x0 * sm.dtw[d0][0] + x1 * sm.dtw[d0][1] + x2 * sm.dtw[d0][2] + x3 * sm.dtw[d0][3]);
      float dt1 = softplus_(sm.dtb[d0 + 1] +
          x0 * sm.dtw[d0 + 1][0] + x1 * sm.dtw[d0 + 1][1] +
          x2 * sm.dtw[d0 + 1][2] + x3 * sm.dtw[d0 + 1][3]);

      float du0 = dt0 * u0, du1 = dt1 * u1;
      float y0 = 0.f, y1 = 0.f;
      const float* Arow0 = sm.Af[d0];
      const float* Arow1 = sm.Af[d0 + 1];
      const float* Brow  = &sm.xdbl[sb][DR_];
      const float* Crow  = &sm.xdbl[sb][DR_ + DS_];
      #pragma unroll
      for (int s = 0; s < DS_; ++s) {
        float dA0 = __expf(dt0 * Arow0[s]);    // A <= 0 -> dA in (0,1]
        float dA1 = __expf(dt1 * Arow1[s]);
        float Bs = Brow[s], Cs = Crow[s];
        h0[s] = h0[s] * dA0 + du0 * Bs;
        h1[s] = h1[s] * dA1 + du1 * Bs;
        y0 += h0[s] * Cs;
        y1 += h1[s] * Cs;
      }
      float z0 = bf2f(sm.Zh[sb][d0]), z1 = bf2f(sm.Zh[sb][d0 + 1]);
      float g0 = (y0 + u0 * sm.Dv[d0])     * (z0 * sigmoid_(z0));
      float g1 = (y1 + u1 * sm.Dv[d0 + 1]) * (z1 * sigmoid_(z1));
      sm.Ypk[sb][sj] = f2bf_pk(g0, g1);
    }
    __syncthreads();

    // --- out GEMM: OUT[16,64] = Y[16,128] @ oww^T -> global (waves 0..3) ---
    if (wave < 4) {
      v8f acc = {};
      #pragma unroll
      for (int kt = 0; kt < 4; ++kt) {
        v16bf av = load_frag(&sm.Ypk[0][0], DI_ / 2, 0,         kt * 16);
        v16bf bv = load_frag(&sm.oww[0][0], DI_ / 2, wave * 16, kt * 16);
        acc = __builtin_amdgcn_wmma_f32_16x16x32_bf16(false, av, false, bv,
                                                      (short)0, acc, false, false);
      }
      int n  = (wave << 4) + (lane & 15);
      int mb = (lane >> 4) * 8;
      #pragma unroll
      for (int r = 0; r < 8; ++r)
        out[(size_t)(b0 + mb + r) * (L_ * W_) + t * W_ + n] = acc[r];
    }
    __syncthreads();   // also fences Ypk/scratch before next-t writes
  }
}

extern "C" void kernel_launch(void* const* d_in, const int* in_sizes, int n_in,
                              void* d_out, int out_size, void* d_ws, size_t ws_size,
                              hipStream_t stream) {
  (void)in_sizes; (void)n_in; (void)out_size; (void)d_ws; (void)ws_size;
  const float* enc      = (const float*)d_in[0];
  const float* dec_w    = (const float*)d_in[1];
  const float* dec_b    = (const float*)d_in[2];
  const float* in_w     = (const float*)d_in[3];
  const float* conv_w   = (const float*)d_in[4];
  const float* conv_b   = (const float*)d_in[5];
  const float* xproj_w  = (const float*)d_in[6];
  const float* dtproj_w = (const float*)d_in[7];
  const float* dtproj_b = (const float*)d_in[8];
  const float* A_log    = (const float*)d_in[9];
  const float* Dp       = (const float*)d_in[10];
  const float* out_w    = (const float*)d_in[11];

  dim3 grid(B_ / NB);       // 128 workgroups, one 16-batch tile each
  dim3 block(NTHREADS);     // 32 waves
  hipLaunchKernelGGL(mamba_text_decoder_fused, grid, block, 0, stream,
                     enc, dec_w, dec_b, in_w, conv_w, conv_b,
                     xproj_w, dtproj_w, dtproj_b, A_log, Dp, out_w,
                     (float*)d_out);
}
